// ResampleModule_75222057222803
// MI455X (gfx1250) — compile-verified, matrix operand order checked by reading, not run on hardware
//
#include <hip/hip_runtime.h>

typedef __attribute__((ext_vector_type(16))) _Float16 v16h;
typedef __attribute__((ext_vector_type(8)))  float    v8f;
typedef __attribute__((ext_vector_type(4)))  unsigned int v4u;
typedef __attribute__((ext_vector_type(8)))  int      v8i;
typedef __attribute__((ext_vector_type(4)))  int      v4i;

#define TB   16            // output tile 16x16 pixels
#define HALO 20            // TB+4 (5x5 dependency: conv 3x3 of sim 3x3)
#define SIMW 18            // TB+2 (sim needed at 3x3 conv taps)
#define CC   32
#define HH   768
#define WW   1024
#define HWSZ (HH*WW)

// dynamic LDS partition (floats): lx 12800 | lsim 2592 | loff 512  => 63616 bytes
#define LX_F   (CC * HALO * HALO)
#define LSIM_F (8 * SIMW * SIMW)
#define LOFF_F (TB * TB * 2)
#define SMEM_BYTES ((LX_F + LSIM_F + LOFF_F) * 4)

__global__ __launch_bounds__(256) void resample_fused(
    const float* __restrict__ x, const float* __restrict__ w,
    const float* __restrict__ bias, float* __restrict__ out)
{
    extern __shared__ float smem[];
    float* lx   = smem;                 // [c][y][x], LDS offset 0 (TDM target)
    float* lsim = smem + LX_F;          // [s][y][x]
    float* loff = smem + LX_F + LSIM_F; // [pixel][2]

    const int tid = threadIdx.x;
    const int gx0 = blockIdx.x * TB;
    const int gy0 = blockIdx.y * TB;
    const int bb  = blockIdx.z;
    const float* xb = x + (size_t)bb * CC * HWSZ;

    // ---------- Phase 1: 20x20x32 halo -> LDS ----------
    // Interior blocks: single Tensor Data Mover DMA (3D tile, strides W and H*W).
    // Border blocks: scalar path with zero padding (preserves conv zero-pad semantics).
    const bool interior = (gx0 >= 2) && (gy0 >= 2) &&
                          (gx0 + TB + 1 < WW) && (gy0 + TB + 1 < HH);
    if (interior) {
        if (tid < 32) {   // wave 0 issues the DMA (TDM ignores EXEC)
            size_t gaddr = (size_t)(const void*)(xb + ((size_t)(gy0 - 2) * WW + (gx0 - 2)));
            v4u g0;
            g0[0] = 1u;                                   // count=1 (valid), gather off
            g0[1] = 0u;                                   // lds_addr = 0 (dynamic LDS base)
            g0[2] = (unsigned)(gaddr & 0xFFFFFFFFu);      // global_addr[31:0]
            g0[3] = (unsigned)((gaddr >> 32) & 0x01FFFFFFu) | (2u << 30);  // addr[56:32] | type=2
            v8i g1;
            g1[0] = (int)(2u << 16);                      // data_size = 4B
            g1[1] = (int)((unsigned)HALO << 16);          // tensor_dim0[15:0]
            g1[2] = (int)((unsigned)HALO << 16);          // dim0 hi | tensor_dim1[15:0]
            g1[3] = (int)((unsigned)HALO << 16);          // dim1 hi | tile_dim0 = 20
            g1[4] = (int)((unsigned)HALO | ((unsigned)CC << 16)); // tile_dim1=20, tile_dim2=32
            g1[5] = (int)WW;                              // tensor_dim0_stride lo32 = W
            g1[6] = (int)(((unsigned)HWSZ & 0xFFFFu) << 16); // stride0 hi16 | stride1[15:0]
            g1[7] = (int)((unsigned)HWSZ >> 16);          // tensor_dim1_stride[47:16]
            v4i g2;
            g2[0] = CC;                                   // tensor_dim2
            g2[1] = 0; g2[2] = 0; g2[3] = 0;
            v4i g3;
            g3[0] = 0; g3[1] = 0; g3[2] = 0; g3[3] = 0;
            v8i gz;                                       // gather-mode index payload (unused)
            gz[0] = 0; gz[1] = 0; gz[2] = 0; gz[3] = 0;
            gz[4] = 0; gz[5] = 0; gz[6] = 0; gz[7] = 0;
            __builtin_amdgcn_tensor_load_to_lds(g0, g1, g2, g3, gz, 0);
            __builtin_amdgcn_s_wait_tensorcnt(0);
        }
    } else {
        for (int c = 0; c < CC; ++c) {
            for (int p = tid; p < HALO * HALO; p += 256) {
                int hy = p / HALO, hx = p % HALO;
                int gy = gy0 + hy - 2, gw = gx0 + hx - 2;
                float v = 0.f;
                if (gy >= 0 && gy < HH && gw >= 0 && gw < WW)
                    v = xb[c * HWSZ + gy * WW + gw];
                lx[c * (HALO * HALO) + p] = v;
            }
        }
    }
    __syncthreads();

    // ---------- Phase 2: cosine similarity on 18x18 region ----------
    for (int sp = tid; sp < SIMW * SIMW; sp += 256) {
        int sy = sp / SIMW, sx = sp % SIMW;
        int cbase = (sy + 1) * HALO + (sx + 1);   // halo coords of center
        float cv[CC];
        float nc2 = 0.f;
        #pragma unroll
        for (int c = 0; c < CC; ++c) { float v = lx[c * 400 + cbase]; cv[c] = v; nc2 += v * v; }
        float normc = sqrtf(nc2);
        int s = 0;
        #pragma unroll
        for (int dy = -1; dy <= 1; ++dy) {
            #pragma unroll
            for (int dx = -1; dx <= 1; ++dx) {
                if (dy == 0 && dx == 0) continue;
                int nb = cbase + dy * HALO + dx;
                float dot = 0.f, nn2 = 0.f;
                #pragma unroll
                for (int c = 0; c < CC; ++c) { float nv = lx[c * 400 + nb]; dot += cv[c] * nv; nn2 += nv * nv; }
                float denom = fmaxf(normc * sqrtf(nn2), 1e-8f);
                lsim[s * (SIMW * SIMW) + sp] = dot / denom;
                ++s;
            }
        }
    }
    __syncthreads();

    // ---------- Phase 3: build B fragments (conv weights), K = 12 steps of 32 ----------
    // K layout (consistent across A and B): k in [0,288): k = tap*32 + ch     (x channels)
    //                                       k in [288,384): k' = tap*10 + c,
    //                                       c<8 -> sim ch 32+c, c==8 -> xx(ch40), c==9 -> yy(ch41)
    const int lane   = tid & 31;
    const int laneHi = lane >> 4;      // 0: lanes 0-15, 1: lanes 16-31
    const int nCol   = lane & 15;      // B column / D column / A row
    const int wv     = tid >> 5;       // wave id 0..7

    v16h bfrag[12];
    #pragma unroll
    for (int t = 0; t < 12; ++t) {
        #pragma unroll
        for (int i = 0; i < 16; ++i) {
            int kk = i + ((i >= 8) ? 8 : 0) + (laneHi ? 8 : 0);   // K row within step
            float val = 0.f;
            if (nCol < 2) {
                if (t < 9) {
                    int ch = kk;                          // x channel
                    const int ki = t / 3, kj = t % 3;
                    val = w[((nCol * 42 + ch) * 3 + ki) * 3 + kj];
                } else {
                    int kp = (t - 9) * 32 + kk;
                    if (kp < 90) {
                        int tap = kp / 10, c = kp % 10;
                        int ch  = (c < 8) ? (32 + c) : (40 + (c - 8));
                        int ki = tap / 3, kj = tap % 3;
                        val = w[((nCol * 42 + ch) * 3 + ki) * 3 + kj];
                    }
                }
            }
            bfrag[t][i] = (_Float16)val;
        }
    }

    // ---------- Phase 4: implicit-GEMM conv via WMMA (each wave: 2 groups of 16 pixels) ----------
    #pragma unroll
    for (int g = 0; g < 2; ++g) {
        const int pbase = wv * 32 + g * 16;   // linear pixel base in tile
        const int py = pbase >> 4;            // tile row (constant per group)
        const int px = nCol;                  // tile col = A-matrix row
        v8f acc = {0.f, 0.f, 0.f, 0.f, 0.f, 0.f, 0.f, 0.f};
        #pragma unroll
        for (int t = 0; t < 12; ++t) {
            v16h a;
            #pragma unroll
            for (int i = 0; i < 16; ++i) {
                int kk = i + ((i >= 8) ? 8 : 0) + (laneHi ? 8 : 0);
                float val = 0.f;
                if (t < 9) {
                    const int dy = t / 3 - 1, dx = t % 3 - 1;
                    int ch = kk;
                    val = lx[ch * 400 + (py + dy + 2) * HALO + (px + dx + 2)];
                } else {
                    int kp = (t - 9) * 32 + kk;
                    if (kp < 90) {
                        int tap = kp / 10, c = kp % 10;
                        int dy = tap / 3 - 1, dx = tap % 3 - 1;
                        if (c < 8) {
                            val = lsim[c * (SIMW * SIMW) + (py + dy + 1) * SIMW + (px + dx + 1)];
                        } else {
                            int gh = gy0 + py + dy, gw2 = gx0 + px + dx;
                            if (gh >= 0 && gh < HH && gw2 >= 0 && gw2 < WW)
                                val = (c == 8) ? (2.f * (float)gh  / (float)(HH - 1) - 1.f)
                                               : (2.f * (float)gw2 / (float)(WW - 1) - 1.f);
                        }
                    }
                }
                a[i] = (_Float16)val;
            }
            acc = __builtin_amdgcn_wmma_f32_16x16x32_f16(
                      false, a, false, bfrag[t], (short)0, acc, false, false);
        }
        // D layout: lane holds column N=nCol; VGPR r -> row M = r + 8*laneHi
        if (nCol < 2) {
            #pragma unroll
            for (int r = 0; r < 8; ++r)
                loff[(pbase + r + laneHi * 8) * 2 + nCol] = acc[r];
        }
    }
    __syncthreads();

    // ---------- Phase 5: bilinear gather (grid_sample with border clamp) ----------
    {
        const int py = tid >> 4, px = tid & 15;
        const int gh = gy0 + py, gw2 = gx0 + px;
        float off0 = loff[tid * 2 + 0] + bias[0];
        float off1 = loff[tid * 2 + 1] + bias[1];
        // ix = clip((gx+1)*W/2 - 0.5) reduces to clip(w + off0, 0, W-1); same for iy
        float ix = fminf(fmaxf((float)gw2 + off0, 0.f), (float)(WW - 1));
        float iy = fminf(fmaxf((float)gh  + off1, 0.f), (float)(HH - 1));
        float x0f = floorf(ix), y0f = floorf(iy);
        float wxf = ix - x0f, wyf = iy - y0f;
        int x0i = (int)x0f, y0i = (int)y0f;
        int x1i = min(x0i + 1, WW - 1), y1i = min(y0i + 1, HH - 1);
        float w00 = (1.f - wxf) * (1.f - wyf), w01 = wxf * (1.f - wyf);
        float w10 = (1.f - wxf) * wyf,         w11 = wxf * wyf;
        int i00 = y0i * WW + x0i, i01 = y0i * WW + x1i;
        int i10 = y1i * WW + x0i, i11 = y1i * WW + x1i;
        float* ob = out + (size_t)bb * CC * HWSZ + (size_t)gh * WW + gw2;
        #pragma unroll 4
        for (int c = 0; c < CC; ++c) {
            const float* bc = xb + (size_t)c * HWSZ;
            float v = w00 * bc[i00] + w01 * bc[i01] + w10 * bc[i10] + w11 * bc[i11];
            ob[(size_t)c * HWSZ] = v;
        }
    }
}

extern "C" void kernel_launch(void* const* d_in, const int* in_sizes, int n_in,
                              void* d_out, int out_size, void* d_ws, size_t ws_size,
                              hipStream_t stream) {
    const float* x    = (const float*)d_in[0];
    const float* w    = (const float*)d_in[1];
    const float* bias = (const float*)d_in[2];
    float* out = (float*)d_out;
    int B = in_sizes[0] / (CC * HWSZ);
    dim3 grid(WW / TB, HH / TB, B);
    resample_fused<<<grid, dim3(256), SMEM_BYTES, stream>>>(x, w, bias, out);
}
